// QuaMap_83356725280795
// MI455X (gfx1250) — matrix operand order repaired and checked by reading, not compiled, not run on hardware
//
#include <hip/hip_runtime.h>
#include <hip/hip_bf16.h>

#define B_   64
#define W_   16
#define N_   196
#define D_   384
#define G_   4
#define V_   96
#define H_   64
#define S_   14
#define KPAD 224   // 7 chunks of 32
#define MPAD 208   // 13 tiles of 16
#define MT_  13
#define LDST 40    // padded LDS row stride (bf16 elems): 80B, 16B-aligned, conflict-friendly

typedef __attribute__((ext_vector_type(16))) __bf16 v16bf;
typedef __attribute__((ext_vector_type(8)))  float  v8f;

// ---- CDNA5 async global->LDS copy (ASYNCcnt) ----
// Builtin prototype (from compiler diagnostic): (v4i AS1*, v4i AS3*, imm int, imm int)
typedef int v4i_ __attribute__((vector_size(16)));
typedef __attribute__((address_space(1))) v4i_* gv4p;
typedef __attribute__((address_space(3))) v4i_* lv4p;

__device__ __forceinline__ void async_copy_b128(const void* gsrc, void* ldst) {
#if __has_builtin(__builtin_amdgcn_global_load_async_to_lds_b128)
    __builtin_amdgcn_global_load_async_to_lds_b128(
        (gv4p)(uintptr_t)gsrc, (lv4p)(uintptr_t)ldst, 0, 0);
#else
    unsigned loff = (unsigned)(uintptr_t)ldst;         // low 32 bits of generic = LDS offset
    unsigned long long ga = (unsigned long long)(uintptr_t)gsrc;
    asm volatile("global_load_async_to_lds_b128 %0, %1, off" :: "v"(loff), "v"(ga) : "memory");
#endif
}
__device__ __forceinline__ void wait_asynccnt0() {
#if __has_builtin(__builtin_amdgcn_s_wait_asynccnt)
    __builtin_amdgcn_s_wait_asynccnt(0);
#else
    asm volatile("s_wait_asynccnt 0x0" ::: "memory");
#endif
}

// ---------------- Kernel 1: pos_proj (H x 5, padded to 8) ----------------
__global__ void pos_proj_kernel(const float* __restrict__ centers,
                                const float* __restrict__ spreads,
                                float* __restrict__ pp) {
    int h = threadIdx.x;
    if (h >= H_) return;
    float S00 = spreads[h*4+0], S01 = spreads[h*4+1];
    float S10 = spreads[h*4+2], S11 = spreads[h*4+3];
    float a = S00*S00 + S01*S01;
    float b = S00*S10 + S01*S11;
    float c = S10*S10 + S11*S11;
    float mu1 = centers[h*2+0], mu2 = centers[h*2+1];
    pp[h*8+0] = a*mu1 + b*mu2;
    pp[h*8+1] = c*mu2 + b*mu1;
    pp[h*8+2] = -0.5f*a;
    pp[h*8+3] = -0.5f*c;
    pp[h*8+4] = -b;
    pp[h*8+5] = 0.f; pp[h*8+6] = 0.f; pp[h*8+7] = 0.f;
}

// ---- Kernel 2: fused rel-index score + softmax -> bf16 weights (padded) ----
__global__ __launch_bounds__(256) void softmax_kernel(const float* __restrict__ pp,
                                                      __bf16* __restrict__ ww) {
    int lane = threadIdx.x & 31;
    int row  = blockIdx.x * 8 + (threadIdx.x >> 5);   // 0 .. H_*MPAD-1
    int h = row / MPAD;
    int m = row - h * MPAD;
    __bf16* dst = ww + (size_t)row * KPAD;
    if (m >= N_) {                       // zero the M padding rows
        #pragma unroll
        for (int i = 0; i < 7; ++i) dst[lane + 32*i] = (__bf16)0.0f;
        return;
    }
    float p0 = pp[h*8+0], p1 = pp[h*8+1], p2 = pp[h*8+2], p3 = pp[h*8+3], p4 = pp[h*8+4];
    int mx = m % S_, my = m / S_;
    float sc[7];
    float lmax = -1e30f;
    #pragma unroll
    for (int i = 0; i < 7; ++i) {
        int n = lane + 32*i;
        if (n < N_) {
            float dx = (float)((n % S_) - mx);
            float dy = (float)((n / S_) - my);
            float sig = 1.0f / (1.0f + __expf(-dx*dy));
            float s = p0*dx + p1*dy + p2*dx*dx + p3*dy*dy + p4*sig;
            sc[i] = s;
            lmax = fmaxf(lmax, s);
        } else sc[i] = -1e30f;
    }
    #pragma unroll
    for (int off = 16; off >= 1; off >>= 1) lmax = fmaxf(lmax, __shfl_xor(lmax, off, 32));
    float lsum = 0.f;
    #pragma unroll
    for (int i = 0; i < 7; ++i) {
        int n = lane + 32*i;
        float e = (n < N_) ? __expf(sc[i] - lmax) : 0.f;
        sc[i] = e;
        lsum += e;
    }
    #pragma unroll
    for (int off = 16; off >= 1; off >>= 1) lsum += __shfl_xor(lsum, off, 32);
    float inv = 1.0f / lsum;
    #pragma unroll
    for (int i = 0; i < 7; ++i) {
        int n = lane + 32*i;
        dst[n] = (__bf16)((n < N_) ? sc[i] * inv : 0.f);
    }
}

// ---------------- Kernel 3: batched GEMM via bf16 WMMA ----------------
// Block = (b, w, mtile): computes out[b,w, m0..m0+15, :] for ALL 4 gamma slices.
// 6 waves; wave j owns N-tile v in [16j,16j+16) and accumulates 4 tiles (one per g).
__global__ __launch_bounds__(192) void gemm_kernel(const float*  __restrict__ x,
                                                   const __bf16* __restrict__ ww,
                                                   const float*  __restrict__ bias,
                                                   float*        __restrict__ out) {
    __shared__ __align__(16) __bf16 Asm[4 * 16 * LDST];   // per-g A tile: [m][k]
    __shared__ __align__(16) __bf16 Bsm[4 * 96 * LDST];   // per-g B tile transposed: [v][k]

    int id = blockIdx.x;
    int mt = id % MT_;
    int w  = (id / MT_) % W_;
    int b  =  id / (MT_ * W_);
    int bw = b * W_ + w;
    int m0 = mt * 16;

    int tid  = threadIdx.x;
    int lane = tid & 31;
    int wave = tid >> 5;          // 0..5
    int vbase = wave * 16;

    const float4* xsrc = (const float4*)(x + (size_t)bw * N_ * D_);  // [n][D/4]
    int stv = tid % 96;           // B staging: column v
    int sth = tid / 96;           // B staging: k-half (0..1)

    v8f acc[4] = {};
    for (int kc = 0; kc < 7; ++kc) {
        int k0 = kc * 32;
        // ---- stage A via async copies: 4 heads x 16 rows x 64B = 256 x 16B ----
        {
            int idx = tid;                      // threads 0..191 -> chunk tid
            int g  = idx >> 6, rem = idx & 63;
            int r  = rem >> 2, q = rem & 3;
            async_copy_b128(ww + ((size_t)(g * W_ + w) * MPAD + m0 + r) * KPAD + k0 + q * 8,
                            &Asm[g * (16 * LDST) + r * LDST + q * 8]);
            if (tid < 64) {                     // chunks 192..255
                int idx2 = 192 + tid;
                int g2 = idx2 >> 6, rem2 = idx2 & 63;
                int r2 = rem2 >> 2, q2 = rem2 & 3;
                async_copy_b128(ww + ((size_t)(g2 * W_ + w) * MPAD + m0 + r2) * KPAD + k0 + q2 * 8,
                                &Asm[g2 * (16 * LDST) + r2 * LDST + q2 * 8]);
            }
        }
        // ---- stage B: x[k0..k0+31][all d], fp32 -> bf16, [g][v][k] in LDS ----
        // A's K-pad columns (>=196) are zero, so clamped (garbage) B rows are harmless.
        union Pack { __bf16 hh[16]; uint4 u[2]; };
        Pack pk[4];
        #pragma unroll
        for (int jg = 0; jg < 2; ++jg) {
            float4 f[8];
            #pragma unroll
            for (int j = 0; j < 8; ++j) {
                int nk = k0 + sth * 16 + jg * 8 + j;
                int nkc = nk < N_ ? nk : N_ - 1;
                f[j] = xsrc[(size_t)nkc * (D_ / 4) + stv];
            }
            #pragma unroll
            for (int j = 0; j < 8; ++j) {
                int jj = jg * 8 + j;
                pk[0].hh[jj] = (__bf16)f[j].x;
                pk[1].hh[jj] = (__bf16)f[j].y;
                pk[2].hh[jj] = (__bf16)f[j].z;
                pk[3].hh[jj] = (__bf16)f[j].w;
            }
        }
        #pragma unroll
        for (int g = 0; g < 4; ++g) {
            uint4* d0 = (uint4*)&Bsm[g * (96 * LDST) + stv * LDST + sth * 16];
            d0[0] = pk[g].u[0];
            d0[1] = pk[g].u[1];
        }
        if (kc < 6) __builtin_prefetch(&xsrc[(size_t)(k0 + 32 + sth * 16) * (D_ / 4) + stv], 0, 1);
        wait_asynccnt0();
        __syncthreads();
        // ---- fragments in contiguous k order (k = 16*(lane/16) + elem) for A and B + 4 WMMAs ----
        int rowm = lane & 15;
        int kA = (lane < 16) ? 0 : 16;   // A: elems 0..15 -> K kA..kA+15
        int kB = (lane < 16) ? 0 : 16;   // B: elems 0..15 -> K kB..kB+15, col = vbase+(lane&15)
        #pragma unroll
        for (int g = 0; g < 4; ++g) {
            union { v16bf v; uint4 u[2]; } af, bf;
            const __bf16* Ag = &Asm[g * (16 * LDST)];
            const __bf16* Bg = &Bsm[g * (96 * LDST)];
            af.u[0] = *(const uint4*)&Ag[rowm * LDST + kA];
            af.u[1] = *(const uint4*)&Ag[rowm * LDST + kA + 8];
            const uint4* bp = (const uint4*)&Bg[(vbase + rowm) * LDST + kB];
            bf.u[0] = bp[0];
            bf.u[1] = bp[1];
            acc[g] = __builtin_amdgcn_wmma_f32_16x16x32_bf16(false, af.v, false, bf.v,
                                                             (short)0, acc[g], false, false);
        }
        __syncthreads();
    }
    // ---- store: D layout: VGPR i -> M = (lane<16 ? i : 8+i), N col = lane&15 ----
    int rowm = lane & 15;
    int v = vbase + rowm;
    float* gO = out + (size_t)bw * N_ * D_ + v * G_;
    const float* b0 = bias + (size_t)(w * G_ + 0) * N_;
    const float* b1 = bias + (size_t)(w * G_ + 1) * N_;
    const float* b2 = bias + (size_t)(w * G_ + 2) * N_;
    const float* b3 = bias + (size_t)(w * G_ + 3) * N_;
    int mhi = (lane < 16) ? 0 : 8;
    #pragma unroll
    for (int i = 0; i < 8; ++i) {
        int m = m0 + mhi + i;
        if (m < N_) {
            float4 o = { acc[0][i] + b0[m], acc[1][i] + b1[m],
                         acc[2][i] + b2[m], acc[3][i] + b3[m] };
            *(float4*)&gO[(size_t)m * D_] = o;
        }
    }
}

extern "C" void kernel_launch(void* const* d_in, const int* in_sizes, int n_in,
                              void* d_out, int out_size, void* d_ws, size_t ws_size,
                              hipStream_t stream) {
    const float* x       = (const float*)d_in[0];
    const float* centers = (const float*)d_in[1];
    const float* spreads = (const float*)d_in[2];
    const float* bias    = (const float*)d_in[3];
    float* out = (float*)d_out;

    float*  pp = (float*)d_ws;                          // H*8 floats
    __bf16* ww = (__bf16*)((char*)d_ws + 2048);         // H*MPAD*KPAD bf16 (~6 MB)

    pos_proj_kernel<<<1, 64, 0, stream>>>(centers, spreads, pp);
    softmax_kernel<<<(H_ * MPAD) / 8, 256, 0, stream>>>(pp, ww);
    gemm_kernel<<<B_ * W_ * MT_, 192, 0, stream>>>(x, ww, bias, out);
}
